// MultiHeadAttention_33560874451472
// MI455X (gfx1250) — matrix operand order repaired and checked by reading, not compiled, not run on hardware
//
#include <hip/hip_runtime.h>
#include <hip/hip_bf16.h>

#ifndef __has_builtin
#define __has_builtin(x) 0
#endif
#define HAS_TDM __has_builtin(__builtin_amdgcn_tensor_load_to_lds)

// ---------------- problem constants ----------------
#define B_   2
#define T_   2048
#define DIN  1024
#define DOUT 1024
#define HN   16
#define DH   64
#define LN10000 9.2103403719761836f

typedef __attribute__((ext_vector_type(16))) _Float16 v16h;
typedef __attribute__((ext_vector_type(8)))  _Float16 v8h;
typedef __attribute__((ext_vector_type(8)))  float    v8f;

#define WMMA16(a, b, c) \
    __builtin_amdgcn_wmma_f32_16x16x32_f16(false, (a), false, (b), (short)0, (c), false, false)

// ---------------- fragment helpers -----------------
__device__ inline v16h frag_combine(v8h lo, v8h hi8) {
    v16h r;
#pragma unroll
    for (int i = 0; i < 8; ++i) { r[i] = lo[i]; r[i + 8] = hi8[i]; }
    return r;
}

// A (16x32 f16) from row-major storage: lane M = lane&15, hi = lane>>4,
// elems 0..7 -> K = kbase+8*hi+e ; elems 8..15 -> K = kbase+16+8*hi+(e-8)
__device__ inline v16h load_frag_a(const _Float16* __restrict__ base, int ld, int m, int kbase, int lane) {
    int row = m + (lane & 15);
    int hi  = lane >> 4;
    const _Float16* p = base + (size_t)row * ld + kbase + 8 * hi;
    return frag_combine(*(const v8h*)(p), *(const v8h*)(p + 16));
}

// B (32x16 f16) from K-major storage base[n][k]: lane N = lane&15, hi = lane>>4,
// elems 0..7 -> K = kbase+8*hi+e ; elems 8..15 -> K = kbase+16+8*hi+(e-8)
__device__ inline v16h load_frag_b(const _Float16* __restrict__ base, int ld, int n, int kbase, int lane) {
    int col = n + (lane & 15);
    int hi  = lane >> 4;
    const _Float16* p = base + (size_t)col * ld + kbase + 8 * hi;
    return frag_combine(*(const v8h*)(p), *(const v8h*)(p + 16));
}

#if HAS_TDM
typedef __attribute__((ext_vector_type(4))) unsigned int u32x4_t;
typedef __attribute__((ext_vector_type(8))) int          i32x8_t;
typedef __attribute__((ext_vector_type(4))) int          i32x4_t;

__device__ inline unsigned lds_off(const void* p) {
    // flat LDS addresses: low 32 bits = byte offset within the WG allocation
    return (unsigned)(unsigned long long)(uintptr_t)p;
}

// TDM: load a (tile_rows x 64) f16 tile (row stride = row_stride_elems in f16 units)
// into LDS at lds_byte, with 8-half padding after every 64 halves -> LDS stride 72.
__device__ inline void tdm_load_tile(const _Float16* gsrc, unsigned lds_byte,
                                     unsigned tile_rows, unsigned long long row_stride_elems) {
    unsigned long long ga = (unsigned long long)(uintptr_t)gsrc;
    u32x4_t g0;
    g0.x = 1u;                                            // count=1, user descriptor
    g0.y = lds_byte;                                      // lds_addr [63:32]
    g0.z = (unsigned)ga;                                  // global_addr [95:64]
    g0.w = (unsigned)((ga >> 32) & 0x01FFFFFFu)           // global_addr [120:96]
         | (2u << 30);                                    // type=2 ("image") [127:126]
    i32x8_t g1;
    g1[0] = (int)((1u << 16)      // data_size = 2 bytes
                | (1u << 20)      // pad_enable
                | (4u << 22)      // pad_interval code 4 => every 32 dwords (64 halves)
                | (3u << 25));    // pad_amount  code 3 => 4 dwords (8 halves)
    g1[1] = (int)(0x0000u << 16);                         // tensor_dim0[15:0] (td0=0x40000000)
    g1[2] = (int)(0x4000u);                               // td0[31:16] ; tensor_dim1[15:0]=0
    g1[3] = (int)(0x4000u | (64u << 16));                 // td1[31:16] ; tile_dim0 = 64
    g1[4] = (int)(tile_rows & 0xFFFFu);                   // tile_dim1 ; tile_dim2 = 0
    g1[5] = (int)(unsigned)(row_stride_elems & 0xFFFFFFFFull);     // dim0_stride[31:0]
    g1[6] = (int)(unsigned)((row_stride_elems >> 32) & 0xFFFFull); // dim0_stride[47:32]
    g1[7] = 0;                                            // dim1_stride = 0 (unused)
    i32x4_t z4 = {0, 0, 0, 0};
    i32x8_t z8 = {0, 0, 0, 0, 0, 0, 0, 0};
    // amdgpu-toolchain (clang-23) 6-arg form: (g0, g1, g2, g3, extra, cpol)
    __builtin_amdgcn_tensor_load_to_lds(g0, g1, z4, z4, z8, 0);
}
#endif

// =====================================================================
// Kernel 1: QKV projection + RoPE.  grid (M/128, N/64, 3), block 256.
// C = x(4096x1024) * W(1024x1024); Q/K get RoPE; stored f16.
// Q,K -> [B*H][T][DH]; V -> transposed [B*H][DH][T]
// =====================================================================
__global__ __launch_bounds__(256)
void qkv_rope_kernel(const float* __restrict__ x,
                     const float* __restrict__ Wq, const float* __restrict__ Wk,
                     const float* __restrict__ Wv,
                     _Float16* __restrict__ qb, _Float16* __restrict__ kb,
                     _Float16* __restrict__ vtb) {
    const int ASTR = 40, BSTR = 40;
    __shared__ __align__(16) _Float16 lA[128 * 40];
    __shared__ __align__(16) _Float16 lB[64 * 40];

    int tid = threadIdx.x;
    int lane = tid & 31, wave = tid >> 5;
    int wm = wave & 3, wn = wave >> 2;          // 4 x 2 wave grid
    int m0 = blockIdx.x * 128;
    int n0 = blockIdx.y * 64;
    int wsel = blockIdx.z;
    const float* W = (wsel == 0) ? Wq : ((wsel == 1) ? Wk : Wv);

    v8f acc[2][2];
#pragma unroll
    for (int i = 0; i < 2; ++i)
#pragma unroll
        for (int j = 0; j < 2; ++j)
#pragma unroll
            for (int r = 0; r < 8; ++r) acc[i][j][r] = 0.0f;

    for (int k0 = 0; k0 < DIN; k0 += 32) {
        __syncthreads();
        // stage A tile 128x32 (f32 -> f16), row-major
#pragma unroll
        for (int j = 0; j < 4; ++j) {
            int id  = tid + 256 * j;
            int row = id >> 3;
            int c4  = (id & 7) * 4;
            const float4 f = *(const float4*)(x + (size_t)(m0 + row) * DIN + k0 + c4);
            if (k0 + 32 < DIN)
                __builtin_prefetch(x + (size_t)(m0 + row) * DIN + k0 + 32 + c4, 0, 1);
            _Float16* d = lA + row * ASTR + c4;
            d[0] = (_Float16)f.x; d[1] = (_Float16)f.y;
            d[2] = (_Float16)f.z; d[3] = (_Float16)f.w;
        }
        // stage B tile 32(k) x 64(n), transposed to K-major [n][k]
#pragma unroll
        for (int j = 0; j < 2; ++j) {
            int id = tid + 256 * j;
            int kk = id >> 4;
            int c4 = (id & 15) * 4;
            const float4 f = *(const float4*)(W + (size_t)(k0 + kk) * DOUT + n0 + c4);
            if (k0 + 32 < DIN)
                __builtin_prefetch(W + (size_t)(k0 + 32 + kk) * DOUT + n0 + c4, 0, 1);
            lB[(c4 + 0) * BSTR + kk] = (_Float16)f.x;
            lB[(c4 + 1) * BSTR + kk] = (_Float16)f.y;
            lB[(c4 + 2) * BSTR + kk] = (_Float16)f.z;
            lB[(c4 + 3) * BSTR + kk] = (_Float16)f.w;
        }
        __syncthreads();

        v16h a0 = load_frag_a(lA, ASTR, wm * 32 + 0,  0, lane);
        v16h a1 = load_frag_a(lA, ASTR, wm * 32 + 16, 0, lane);
        v16h b0 = load_frag_b(lB, BSTR, wn * 32 + 0,  0, lane);
        v16h b1 = load_frag_b(lB, BSTR, wn * 32 + 16, 0, lane);
        acc[0][0] = WMMA16(a0, b0, acc[0][0]);
        acc[0][1] = WMMA16(a0, b1, acc[0][1]);
        acc[1][0] = WMMA16(a1, b0, acc[1][0]);
        acc[1][1] = WMMA16(a1, b1, acc[1][1]);
    }

    // epilogue: RoPE (Q,K) + f16 store
    int hi = lane >> 4, ln = lane & 15;
#pragma unroll
    for (int i = 0; i < 2; ++i) {
#pragma unroll
        for (int j = 0; j < 2; ++j) {
            int N  = n0 + wn * 32 + j * 16 + ln;
            int h  = N >> 6, dh = N & 63;
            float vals[8];
#pragma unroll
            for (int r = 0; r < 8; ++r) vals[r] = acc[i][j][r];

            if (wsel < 2) {
                // reference: inv_freq[i] = 10000^(-2*(2i)/64) = exp(-i*ln(1e4)/16)
                int   i2  = dh >> 1;
                float inv = __expf(-(float)i2 * (LN10000 / 16.0f));
                bool  even = ((N & 1) == 0);
#pragma unroll
                for (int r = 0; r < 8; ++r) {
                    int M = m0 + wm * 32 + i * 16 + 8 * hi + r;
                    int t = M & (T_ - 1);
                    float ang = (float)t * inv;
                    float c = __cosf(ang), s = __sinf(ang);
                    float partner = __shfl_xor(vals[r], 1);
                    vals[r] = even ? (vals[r] * c - partner * s)
                                   : (partner * s + vals[r] * c);
                }
            }
#pragma unroll
            for (int r = 0; r < 8; ++r) {
                int M = m0 + wm * 32 + i * 16 + 8 * hi + r;
                int b = M >> 11;            // /T_
                int t = M & (T_ - 1);
                int bh = b * HN + h;
                if (wsel == 0)
                    qb[((size_t)bh * T_ + t) * DH + dh] = (_Float16)vals[r];
                else if (wsel == 1)
                    kb[((size_t)bh * T_ + t) * DH + dh] = (_Float16)vals[r];
                else
                    vtb[((size_t)bh * DH + dh) * T_ + t] = (_Float16)vals[r];
            }
        }
    }
}

// =====================================================================
// Kernel 2: causal flash attention. grid (T/64, B*H), block 128 (4 waves).
// Each wave: 16 queries; block processes key chunks of 64.
// TDM path: K/V tiles DMA'd to LDS, double buffered, TENSORcnt-gated.
// ctx stored f16 in [B][T][H*DH] layout (ready for the output GEMM).
// =====================================================================
__global__ __launch_bounds__(128)
void attn_kernel(const _Float16* __restrict__ qb, const _Float16* __restrict__ kb,
                 const _Float16* __restrict__ vtb, _Float16* __restrict__ ctx) {
    __shared__ __align__(16) _Float16 lP[4][16 * 72];
#if HAS_TDM
    __shared__ __align__(16) _Float16 kTile[2][64 * 72];   // keys x dh, padded stride 72
    __shared__ __align__(16) _Float16 vTile[2][64 * 72];   // dh x keys, padded stride 72
#endif

    int lane = threadIdx.x & 31, wave = threadIdx.x >> 5;
    int hi = lane >> 4, ln = lane & 15;
    int bh = blockIdx.y;
    int q0 = (blockIdx.x * 4 + wave) * 16;

    const _Float16* Qh = qb  + (size_t)bh * T_ * DH;
    const _Float16* Kh = kb  + (size_t)bh * T_ * DH;
    const _Float16* Vh = vtb + (size_t)bh * DH * T_;
    _Float16* pbuf = lP[wave];

    v16h qa0 = load_frag_a(Qh, DH, q0, 0,  lane);
    v16h qa1 = load_frag_a(Qh, DH, q0, 32, lane);

    v8f acc[4];
#pragma unroll
    for (int f = 0; f < 4; ++f)
#pragma unroll
        for (int r = 0; r < 8; ++r) acc[f][r] = 0.0f;

    float mrow[8], lrow[8];
#pragma unroll
    for (int r = 0; r < 8; ++r) { mrow[r] = -1e30f; lrow[r] = 0.0f; }

    const float scale = 0.125f;          // 1/sqrt(64)
    const int nchunks = blockIdx.x + 1;  // block covers keys [0, qblock+64)

#if HAS_TDM
    unsigned kOff0 = lds_off(&kTile[0][0]), kOff1 = lds_off(&kTile[1][0]);
    unsigned vOff0 = lds_off(&vTile[0][0]), vOff1 = lds_off(&vTile[1][0]);
    if (wave == 0) {
        tdm_load_tile(Kh,            kOff0, 64, 64);   // chunk 0 keys, contiguous rows
        tdm_load_tile(Vh,            vOff0, 64, T_);   // chunk 0 cols of Vt
        if (nchunks > 1) {
            tdm_load_tile(Kh + 64 * DH, kOff1, 64, 64);
            tdm_load_tile(Vh + 64,      vOff1, 64, T_);
        }
    }
#endif

    for (int c = 0; c < nchunks; ++c) {
        int n0 = c * 64;
#if HAS_TDM
        if (wave == 0) {
            if (c + 1 < nchunks) __builtin_amdgcn_s_wait_tensorcnt(2); // oldest K/V pair done
            else                 __builtin_amdgcn_s_wait_tensorcnt(0);
        }
        __syncthreads();
        const _Float16* Kb = kTile[c & 1];
        const _Float16* Vb = vTile[c & 1];
#endif
        // ---- S = Q K^T over 64 keys (4 S tiles) ----
        v8f s[4];
#pragma unroll
        for (int f = 0; f < 4; ++f)
#pragma unroll
            for (int r = 0; r < 8; ++r) s[f][r] = 0.0f;

#pragma unroll
        for (int f = 0; f < 4; ++f) {
#if HAS_TDM
            v16h kf0 = load_frag_b(Kb, 72, f * 16, 0,  lane);
            v16h kf1 = load_frag_b(Kb, 72, f * 16, 32, lane);
#else
            v16h kf0 = load_frag_b(Kh, DH, n0 + f * 16, 0,  lane);
            v16h kf1 = load_frag_b(Kh, DH, n0 + f * 16, 32, lane);
#endif
            s[f] = WMMA16(qa0, kf0, s[f]);
            s[f] = WMMA16(qa1, kf1, s[f]);
        }

        // ---- online softmax over the 64-key chunk ----
        int c0 = n0 + ln;
#pragma unroll
        for (int r = 0; r < 8; ++r) {
            int tq = q0 + 8 * hi + r;
            float e[4];
            float mx = -1e30f;
#pragma unroll
            for (int f = 0; f < 4; ++f) {
                float v = s[f][r] * scale;
                if (c0 + 16 * f > tq) v = -1e30f;      // causal mask
                e[f] = v;
                mx = fmaxf(mx, v);
            }
            mx = fmaxf(mx, __shfl_xor(mx, 1));
            mx = fmaxf(mx, __shfl_xor(mx, 2));
            mx = fmaxf(mx, __shfl_xor(mx, 4));
            mx = fmaxf(mx, __shfl_xor(mx, 8));
            float nm   = fmaxf(mrow[r], mx);
            float corr = __expf(mrow[r] - nm);
            mrow[r] = nm;

            float rs = 0.0f;
#pragma unroll
            for (int f = 0; f < 4; ++f) {
                float p = __expf(e[f] - nm);
                rs += p;
                pbuf[(8 * hi + r) * 72 + ln + 16 * f] = (_Float16)p;  // C->A layout via LDS
            }
            rs += __shfl_xor(rs, 1);
            rs += __shfl_xor(rs, 2);
            rs += __shfl_xor(rs, 4);
            rs += __shfl_xor(rs, 8);
            lrow[r] = lrow[r] * corr + rs;
#pragma unroll
            for (int f = 0; f < 4; ++f) acc[f][r] *= corr;
        }

        asm volatile("s_wait_dscnt 0" ::: "memory");
        v16h pa0 = load_frag_a(pbuf, 72, 0, 0,  lane);
        v16h pa1 = load_frag_a(pbuf, 72, 0, 32, lane);

        // ---- ctx += P V ----
#pragma unroll
        for (int f = 0; f < 4; ++f) {
#if HAS_TDM
            v16h vb0 = load_frag_b(Vb, 72, f * 16, 0,  lane);
            v16h vb1 = load_frag_b(Vb, 72, f * 16, 32, lane);
#else
            v16h vb0 = load_frag_b(Vh, T_, f * 16, n0,      lane);
            v16h vb1 = load_frag_b(Vh, T_, f * 16, n0 + 32, lane);
#endif
            acc[f] = WMMA16(pa0, vb0, acc[f]);
            acc[f] = WMMA16(pa1, vb1, acc[f]);
        }

#if HAS_TDM
        __syncthreads();   // all waves done with tile (c&1) before DMA overwrite
        if (wave == 0 && c + 2 < nchunks) {
            int nn = (c + 2) * 64;
            tdm_load_tile(Kh + (size_t)nn * DH, (c & 1) ? kOff1 : kOff0, 64, 64);
            tdm_load_tile(Vh + nn,              (c & 1) ? vOff1 : vOff0, 64, T_);
        }
#endif
    }

    int b = bh >> 4, h = bh & 15;
#pragma unroll
    for (int f = 0; f < 4; ++f) {
        int dh = f * 16 + ln;
#pragma unroll
        for (int r = 0; r < 8; ++r) {
            int t = q0 + 8 * hi + r;
            float val = acc[f][r] / lrow[r];
            ctx[((size_t)(b * T_ + t)) * DOUT + h * DH + dh] = (_Float16)val;
        }
    }
}

// =====================================================================
// Kernel 3: output projection: out = ctx(f16) @ Wo + bo, f32 out.
// grid (M/128, N/64), block 256. A frags direct from global (L2-resident).
// =====================================================================
__global__ __launch_bounds__(256)
void out_proj_kernel(const _Float16* __restrict__ ctx, const float* __restrict__ Wo,
                     const float* __restrict__ bo, float* __restrict__ out) {
    const int BSTR = 40;
    __shared__ __align__(16) _Float16 lB[64 * 40];

    int tid = threadIdx.x, lane = tid & 31, wave = tid >> 5;
    int wm = wave & 3, wn = wave >> 2;
    int m0 = blockIdx.x * 128, n0 = blockIdx.y * 64;

    v8f acc[2][2];
#pragma unroll
    for (int i = 0; i < 2; ++i)
#pragma unroll
        for (int j = 0; j < 2; ++j)
#pragma unroll
            for (int r = 0; r < 8; ++r) acc[i][j][r] = 0.0f;

    for (int k0 = 0; k0 < DOUT; k0 += 32) {
        __syncthreads();
#pragma unroll
        for (int j = 0; j < 2; ++j) {
            int id = tid + 256 * j;
            int kk = id >> 4;
            int c4 = (id & 15) * 4;
            const float4 f = *(const float4*)(Wo + (size_t)(k0 + kk) * DOUT + n0 + c4);
            if (k0 + 32 < DOUT)
                __builtin_prefetch(Wo + (size_t)(k0 + 32 + kk) * DOUT + n0 + c4, 0, 1);
            lB[(c4 + 0) * BSTR + kk] = (_Float16)f.x;
            lB[(c4 + 1) * BSTR + kk] = (_Float16)f.y;
            lB[(c4 + 2) * BSTR + kk] = (_Float16)f.z;
            lB[(c4 + 3) * BSTR + kk] = (_Float16)f.w;
        }
        __syncthreads();

        v16h a0 = load_frag_a(ctx, DOUT, m0 + wm * 32,      k0, lane);
        v16h a1 = load_frag_a(ctx, DOUT, m0 + wm * 32 + 16, k0, lane);
        v16h b0 = load_frag_b(lB, BSTR, wn * 32,      0, lane);
        v16h b1 = load_frag_b(lB, BSTR, wn * 32 + 16, 0, lane);
        acc[0][0] = WMMA16(a0, b0, acc[0][0]);
        acc[0][1] = WMMA16(a0, b1, acc[0][1]);
        acc[1][0] = WMMA16(a1, b0, acc[1][0]);
        acc[1][1] = WMMA16(a1, b1, acc[1][1]);
    }

    int hi = lane >> 4, ln = lane & 15;
#pragma unroll
    for (int i = 0; i < 2; ++i) {
#pragma unroll
        for (int j = 0; j < 2; ++j) {
            int N = n0 + wn * 32 + j * 16 + ln;
            float bias = bo[N];
#pragma unroll
            for (int r = 0; r < 8; ++r) {
                int M = m0 + wm * 32 + i * 16 + 8 * hi + r;
                out[(size_t)M * DOUT + N] = acc[i][j][r] + bias;
            }
        }
    }
}

// =====================================================================
extern "C" void kernel_launch(void* const* d_in, const int* in_sizes, int n_in,
                              void* d_out, int out_size, void* d_ws, size_t ws_size,
                              hipStream_t stream) {
    (void)in_sizes; (void)n_in; (void)out_size; (void)ws_size;
    const float* x  = (const float*)d_in[0];
    const float* Wq = (const float*)d_in[1];
    const float* Wk = (const float*)d_in[2];
    const float* Wv = (const float*)d_in[3];
    const float* Wo = (const float*)d_in[4];
    const float* bo = (const float*)d_in[5];
    float* out = (float*)d_out;

    const size_t HEADS_ELEMS = (size_t)B_ * HN * T_ * DH;   // 4,194,304
    _Float16* qbuf = (_Float16*)d_ws;
    _Float16* kbuf = qbuf + HEADS_ELEMS;
    _Float16* vtb  = kbuf + HEADS_ELEMS;
    _Float16* ctx  = vtb  + HEADS_ELEMS;                    // total 32 MB f16

    qkv_rope_kernel<<<dim3((B_ * T_) / 128, DOUT / 64, 3), 256, 0, stream>>>(
        x, Wq, Wk, Wv, qbuf, kbuf, vtb);
    attn_kernel<<<dim3(T_ / 64, B_ * HN), 128, 0, stream>>>(qbuf, kbuf, vtb, ctx);
    out_proj_kernel<<<dim3((B_ * T_) / 128, DOUT / 64), 256, 0, stream>>>(ctx, Wo, bo, out);
}
